// STGNN_7035156431050
// MI455X (gfx1250) — matrix operand order repaired and checked by reading, not compiled, not run on hardware
//
#include <hip/hip_runtime.h>

typedef __attribute__((ext_vector_type(16))) _Float16 v16h;
typedef __attribute__((ext_vector_type(8)))  float    v8f;

#define NT 256

// ---------------- elementwise helpers ----------------
__global__ void k_fill(float* __restrict__ p, float v, long n) {
  long i = (long)blockIdx.x * blockDim.x + threadIdx.x;
  if (i < n) p[i] = v;
}

__global__ void k_deg(const long long* __restrict__ dst, float* __restrict__ deg, long E) {
  long e = (long)blockIdx.x * blockDim.x + threadIdx.x;
  if (e < E) atomicAdd(&deg[dst[e]], 1.0f);
}

__global__ void k_dinv(float* __restrict__ dinv, int N) {
  int i = blockIdx.x * blockDim.x + threadIdx.x;
  if (i < N) { float d = dinv[i]; dinv[i] = (d > 0.f) ? rsqrtf(d) : 0.f; }
}

// Pre-swizzle W[K,OUT] (f32, row-major) into B-fragment order:
// element t -> chunk = t/(512*NCT), ct = (t>>9)%NCT, lane = (t>>4)&31, j = t&15
// holds W[k = chunk*32 + (lane>=16 ? 8:0) + (j<8 ? j : j+8)][n = ct*16 + (lane&15)] as f16,
// so each lane's 16 B-halves for a (chunk, ct) are 32 contiguous bytes.
__global__ void k_swizzle_w(const float* __restrict__ W, _Float16* __restrict__ Wz,
                            int K, int OUT, int total) {
  int t = blockIdx.x * blockDim.x + threadIdx.x;
  if (t >= total) return;
  const int NCT  = OUT >> 4;
  const int j    = t & 15;
  const int lane = (t >> 4) & 31;
  const int ct   = (t >> 9) % NCT;
  const int chnk = t / (NCT << 9);
  const int k    = chnk * 32 + ((lane >= 16) ? 8 : 0) + (j < 8 ? j : j + 8);
  const int n    = ct * 16 + (lane & 15);
  Wz[t] = (k < K) ? (_Float16)W[k * OUT + n] : (_Float16)0.f;
}

// ---------------- WMMA GEMM: C[N,OUT] = H[N,K] @ W[K,OUT] ----------------
// one wave per 16-row tile; all OUT/16 column tiles per wave (A reuse);
// 2 waves (2 row tiles) per block. N divisible by 32.
template<int K, int OUT>
__global__ void k_gemm_wmma(const float* __restrict__ H, const _Float16* __restrict__ Wz,
                            float* __restrict__ C, int N) {
  constexpr int NCT    = OUT / 16;
  constexpr int NCHUNK = (K + 31) / 32;
  const int lane = threadIdx.x & 31;
  const int wave = threadIdx.x >> 5;
  const int col  = lane & 15;
  const bool hi  = lane >= 16;
  const int rt = blockIdx.x * 2 + wave;   // row tile
  const int m  = rt * 16 + col;           // A row supplied by this lane

  v8f c[NCT];
#pragma unroll
  for (int ct = 0; ct < NCT; ++ct) c[ct] = v8f{};

#pragma unroll
  for (int chnk = 0; chnk < NCHUNK; ++chnk) {
    const int kk = chnk * 32;
    // ---- A fragment (16-bit A 16x32 layout) ----
    v16h a;
    if constexpr (K >= 32) {
      // lo lanes: K kk+0..7 & kk+16..23 ; hi lanes: kk+8..15 & kk+24..31
      const float* hrow = H + (size_t)m * K + kk;
      const int b1 = hi ? 8 : 0;
      const int b2 = hi ? 24 : 16;
      const float4 p0 = *(const float4*)(hrow + b1);
      const float4 p1 = *(const float4*)(hrow + b1 + 4);
      const float4 p2 = *(const float4*)(hrow + b2);
      const float4 p3 = *(const float4*)(hrow + b2 + 4);
      a[0]=(_Float16)p0.x;  a[1]=(_Float16)p0.y;  a[2]=(_Float16)p0.z;  a[3]=(_Float16)p0.w;
      a[4]=(_Float16)p1.x;  a[5]=(_Float16)p1.y;  a[6]=(_Float16)p1.z;  a[7]=(_Float16)p1.w;
      a[8]=(_Float16)p2.x;  a[9]=(_Float16)p2.y;  a[10]=(_Float16)p2.z; a[11]=(_Float16)p2.w;
      a[12]=(_Float16)p3.x; a[13]=(_Float16)p3.y; a[14]=(_Float16)p3.z; a[15]=(_Float16)p3.w;
    } else {
      // small-K path (layer 1, K=4): guarded scalar loads, rest zero
#pragma unroll
      for (int i = 0; i < 8; ++i) {
        const int kaBase = (i < 4 ? 2 * i : 16 + 2 * (i - 4)) + (hi ? 8 : 0);
#pragma unroll
        for (int h = 0; h < 2; ++h) {
          const int ka = kk + kaBase + h;
          a[2 * i + h] = (ka < K) ? (_Float16)H[(size_t)m * K + ka] : (_Float16)0.f;
        }
      }
    }
    // ---- B fragments: pre-swizzled, 32 contiguous bytes per lane ----
#pragma unroll
    for (int ct = 0; ct < NCT; ++ct) {
      const v16h b = *(const v16h*)(Wz + (((size_t)(chnk * NCT + ct) << 9) + (lane << 4)));
      c[ct] = __builtin_amdgcn_wmma_f32_16x16x32_f16(false, a, false, b, (short)0, c[ct],
                                                     false, false);
    }
  }
  // C/D layout: VGPR r -> row rt*16 + r (lanes 0-15) / +8 (lanes 16-31)
  const int rbase = rt * 16 + (hi ? 8 : 0);
#pragma unroll
  for (int ct = 0; ct < NCT; ++ct) {
    const int n = ct * 16 + col;
#pragma unroll
    for (int r = 0; r < 8; ++r)
      C[(size_t)(rbase + r) * OUT + n] = c[ct][r];
  }
}

// ---------------- edge gather * norm -> scatter-add ----------------
// thread t: edge e = t >> gshift, feature group g = t & mask (4 floats).
__global__ void k_scatter(const float* __restrict__ HW, const long long* __restrict__ src,
                          const long long* __restrict__ dst, const float* __restrict__ dinv,
                          float* __restrict__ acc, long E, int gshift) {
  long t = (long)blockIdx.x * blockDim.x + threadIdx.x;
  long e = t >> gshift;
  if (e >= E) return;
  const int g   = (int)(t & ((1 << gshift) - 1));
  const int OUT = 4 << gshift;
  const long s = src[e];
  const long d = dst[e];
  const float nrm = dinv[s] * dinv[d];
  const float4 v = *(const float4*)(HW + (size_t)s * OUT + g * 4);
  float* o = acc + (size_t)d * OUT + g * 4;
  atomicAdd(o + 0, v.x * nrm);
  atomicAdd(o + 1, v.y * nrm);
  atomicAdd(o + 2, v.z * nrm);
  atomicAdd(o + 3, v.w * nrm);
}

// ---------------- epilogue: + self-loop term + bias (+ relu) ----------------
__global__ void k_finish(const float* __restrict__ acc, const float* __restrict__ HW,
                         const float* __restrict__ dinv, const float* __restrict__ bias,
                         float* __restrict__ out, long total, int oshift, int relu) {
  long t = (long)blockIdx.x * blockDim.x + threadIdx.x;
  if (t >= total) return;
  const long node = t >> oshift;
  const int  f    = (int)(t & ((1 << oshift) - 1));
  const float di = dinv[node];
  float v = acc[t] + di * di * HW[t] + bias[f];
  if (relu) v = fmaxf(v, 0.f);
  out[t] = v;
}

extern "C" void kernel_launch(void* const* d_in, const int* in_sizes, int n_in,
                              void* d_out, int out_size, void* d_ws, size_t ws_size,
                              hipStream_t stream) {
  const float*     x   = (const float*)d_in[0];
  const long long* ei  = (const long long*)d_in[1];  // int64 [2, E]
  const float*     W1  = (const float*)d_in[2];
  const float*     vb1 = (const float*)d_in[3];
  const float*     W2  = (const float*)d_in[4];
  const float*     vb2 = (const float*)d_in[5];
  const float*     W3  = (const float*)d_in[6];
  const float*     vb3 = (const float*)d_in[7];

  const int  N = in_sizes[0] / 4;       // [N,4] features
  const long E = (long)in_sizes[1] / 2; // edge_index flat = 2*E
  const long long* src = ei;
  const long long* dst = ei + E;

  char* w = (char*)d_ws;
  size_t off = 0;
  auto carve = [&](size_t bytes) -> void* {
    void* p = w + off;
    off = (off + bytes + 255) & ~(size_t)255;
    return p;
  };
  float*    dinv = (float*)carve((size_t)N * 4);
  float*    bufA = (float*)carve((size_t)N * 64 * 4);  // layer activations
  float*    bufB = (float*)carve((size_t)N * 64 * 4);  // h @ W result
  float*    acc  = (float*)carve((size_t)N * 64 * 4);  // scatter accumulator
  // swizzled weights: NCHUNK * NCT * 512 halves each
  _Float16* Wz1 = (_Float16*)carve((size_t)1 * 2 * 512 * 2);
  _Float16* Wz2 = (_Float16*)carve((size_t)1 * 4 * 512 * 2);
  _Float16* Wz3 = (_Float16*)carve((size_t)2 * 4 * 512 * 2);
  float* out = (float*)d_out;

  // --- degrees (self-loop contributes 1 to every node) -> dinv in place ---
  k_fill<<<(N + NT - 1) / NT, NT, 0, stream>>>(dinv, 1.0f, (long)N);
  k_deg <<<(int)((E + NT - 1) / NT), NT, 0, stream>>>(dst, dinv, E);
  k_dinv<<<(N + NT - 1) / NT, NT, 0, stream>>>(dinv, N);

  // --- weights -> f16, B-fragment swizzled ---
  k_swizzle_w<<<4,  NT, 0, stream>>>(W1, Wz1, 4,  32, 1 * 2 * 512);
  k_swizzle_w<<<8,  NT, 0, stream>>>(W2, Wz2, 32, 64, 1 * 4 * 512);
  k_swizzle_w<<<16, NT, 0, stream>>>(W3, Wz3, 64, 64, 2 * 4 * 512);

  // --- Layer 1: x[N,4] @ W1 -> 32, scatter, +self-loop +b1, relu ---
  k_gemm_wmma<4, 32><<<N / 32, 64, 0, stream>>>(x, Wz1, bufB, N);
  k_fill<<<(int)(((long)N * 32 + NT - 1) / NT), NT, 0, stream>>>(acc, 0.f, (long)N * 32);
  k_scatter<<<(int)(((E << 3) + NT - 1) / NT), NT, 0, stream>>>(bufB, src, dst, dinv, acc, E, 3);
  k_finish<<<(int)(((long)N * 32 + NT - 1) / NT), NT, 0, stream>>>(acc, bufB, dinv, vb1, bufA,
                                                                   (long)N * 32, 5, 1);

  // --- Layer 2: h1[N,32] @ W2 -> 64 ---
  k_gemm_wmma<32, 64><<<N / 32, 64, 0, stream>>>(bufA, Wz2, bufB, N);
  k_fill<<<(int)(((long)N * 64 + NT - 1) / NT), NT, 0, stream>>>(acc, 0.f, (long)N * 64);
  k_scatter<<<(int)(((E << 4) + NT - 1) / NT), NT, 0, stream>>>(bufB, src, dst, dinv, acc, E, 4);
  k_finish<<<(int)(((long)N * 64 + NT - 1) / NT), NT, 0, stream>>>(acc, bufB, dinv, vb2, bufA,
                                                                   (long)N * 64, 6, 1);

  // --- Layer 3: h2[N,64] @ W3 -> 64, scatter directly into d_out, no relu ---
  k_gemm_wmma<64, 64><<<N / 32, 64, 0, stream>>>(bufA, Wz3, bufB, N);
  k_fill<<<(int)(((long)N * 64 + NT - 1) / NT), NT, 0, stream>>>(out, 0.f, (long)N * 64);
  k_scatter<<<(int)(((E << 4) + NT - 1) / NT), NT, 0, stream>>>(bufB, src, dst, dinv, out, E, 4);
  k_finish<<<(int)(((long)N * 64 + NT - 1) / NT), NT, 0, stream>>>(out, bufB, dinv, vb3, out,
                                                                   (long)N * 64, 6, 0);
}
